// QuantLinear_30279519437454
// MI455X (gfx1250) — 8-bit matrix operand order repaired and checked by reading, compiled, not run on hardware
//
#include <hip/hip_runtime.h>
#include <hip/hip_bf16.h>
#include <stdint.h>

typedef __attribute__((ext_vector_type(8))) int v8i;

#define DIN  4096
#define DOUT 4096
#define NTOK 8192
#define LDSS 80

__global__ __launch_bounds__(256)
void wquant_kernel(const float* __restrict__ W,
                   int8_t* __restrict__ Wq,
                   float* __restrict__ row_scales,
                   int* __restrict__ w_rowsum) {
    __shared__ float sred[256];
    __shared__ int   sredi[256];
    __shared__ float s_scale;
    const int o = blockIdx.x;
    const int t = threadIdx.x;
    const float* wrow = W + (size_t)o * DIN;

    float amax = 0.f;
    float vals[16];
#pragma unroll
    for (int j = 0; j < 16; ++j) {
        float w = wrow[t + j * 256];
        vals[j] = w;
        amax = fmaxf(amax, fabsf(w));
    }
    sred[t] = amax;
    __syncthreads();
    for (int s = 128; s > 0; s >>= 1) {
        if (t < s) sred[t] = fmaxf(sred[t], sred[t + s]);
        __syncthreads();
    }
    if (t == 0) {
        float sc = fmaxf(sred[0], 1e-8f) / 127.f;
        s_scale = sc;
        row_scales[o] = sc;
    }
    __syncthreads();
    const float inv = 1.f / s_scale;
    int rsum = 0;
#pragma unroll
    for (int j = 0; j < 16; ++j) {
        float q = rintf(vals[j] * inv);
        q = fminf(fmaxf(q, -127.f), 127.f);
        int qi = (int)q;
        rsum += qi;
        Wq[(size_t)o * DIN + t + j * 256] = (int8_t)qi;
    }
    sredi[t] = rsum;
    __syncthreads();
    for (int s = 128; s > 0; s >>= 1) {
        if (t < s) sredi[t] += sredi[t + s];
        __syncthreads();
    }
    if (t == 0) w_rowsum[o] = sredi[0];
}

__global__ __launch_bounds__(256)
void xquant_kernel(const float* __restrict__ X,
                   const float* __restrict__ cmin,
                   const float* __restrict__ cmax,
                   int8_t* __restrict__ Xq,
                   float* __restrict__ col_scales,
                   int* __restrict__ zps) {
    __shared__ float smin[256];
    __shared__ float smax[256];
    __shared__ float s_cs, s_zp;
    const int n = blockIdx.x;
    const int t = threadIdx.x;
    const float* xrow = X + (size_t)n * DIN;

    float mn = 3.4e38f, mx = -3.4e38f;
    float vals[16];
#pragma unroll
    for (int j = 0; j < 16; ++j) {
        int k = t + j * 256;
        float v = xrow[k];
        v = fminf(fmaxf(v, cmin[k]), cmax[k]);
        vals[j] = v;
        mn = fminf(mn, v);
        mx = fmaxf(mx, v);
    }
    smin[t] = mn; smax[t] = mx;
    __syncthreads();
    for (int s = 128; s > 0; s >>= 1) {
        if (t < s) {
            smin[t] = fminf(smin[t], smin[t + s]);
            smax[t] = fmaxf(smax[t], smax[t + s]);
        }
        __syncthreads();
    }
    if (t == 0) {
        float cs = fmaxf(smax[0] - smin[0], 1e-8f) / 255.f;
        float zp = -128.f - rintf(smin[0] / cs);
        s_cs = cs; s_zp = zp;
        col_scales[n] = cs;
        zps[n] = (int)zp;
    }
    __syncthreads();
    const float inv = 1.f / s_cs;
    const float zp = s_zp;
#pragma unroll
    for (int j = 0; j < 16; ++j) {
        float q = rintf(vals[j] * inv) + zp;
        q = fminf(fmaxf(q, -128.f), 127.f);
        Xq[(size_t)n * DIN + t + j * 256] = (int8_t)q;
    }
}

__global__ __launch_bounds__(256)
void gemm_kernel(const int8_t* __restrict__ Xq,
                 const int8_t* __restrict__ Wq,
                 const float* __restrict__ row_scales,
                 const int* __restrict__ w_rowsum,
                 const float* __restrict__ col_scales,
                 const int* __restrict__ zps,
                 const float* __restrict__ bias,
                 float* __restrict__ Y) {
    __shared__ __align__(16) int8_t ldsA[2][128 * LDSS];
    __shared__ __align__(16) int8_t ldsB[2][128 * LDSS];

    const int tid  = threadIdx.x;
    const int lane = tid & 31;
    const int wid  = tid >> 5;
    const int h    = lane >> 4;
    const int ln   = lane & 15;
    const int tileT = blockIdx.x * 128;
    const int tileO = blockIdx.y * 128;
    const int waveT = (wid & 3) * 32;
    const int waveO = (wid >> 2) * 64;

    v8i vzero = {};
    v8i acc[2][4];
#pragma unroll
    for (int i = 0; i < 2; ++i)
#pragma unroll
        for (int j = 0; j < 4; ++j) acc[i][j] = vzero;

    auto stage = [&](int kb, int p) {
#pragma unroll
        for (int i = 0; i < 2; ++i) {
            int c   = tid + i * 256;
            int row = c >> 2;
            int col = (c & 3) * 16;
            uint64_t ga = (uint64_t)(uintptr_t)(Xq + (size_t)(tileT + row) * DIN + kb + col);
            uint32_t la = (uint32_t)(uintptr_t)&ldsA[p][row * LDSS + col];
            asm volatile("global_load_async_to_lds_b128 %0, %1, off"
                         :: "v"(la), "v"(ga) : "memory");
            uint64_t gb = (uint64_t)(uintptr_t)(Wq + (size_t)(tileO + row) * DIN + kb + col);
            uint32_t lb = (uint32_t)(uintptr_t)&ldsB[p][row * LDSS + col];
            asm volatile("global_load_async_to_lds_b128 %0, %1, off"
                         :: "v"(lb), "v"(gb) : "memory");
        }
    };

    stage(0, 0);

#pragma unroll 1
    for (int it = 0; it < DIN / 64; ++it) {
        const int p = it & 1;
        if (it + 1 < DIN / 64) {
            stage((it + 1) * 64, p ^ 1);
            asm volatile("s_wait_asynccnt 4" ::: "memory");
        } else {
            asm volatile("s_wait_asynccnt 0" ::: "memory");
        }
        __syncthreads();

        v8i afrag[2], bfrag[4];
#pragma unroll
        for (int i = 0; i < 2; ++i) {
            const int8_t* ar = &ldsA[p][(waveT + i * 16 + ln) * LDSS + h * 8];
            int2 d0 = *(const int2*)(ar);
            int2 d1 = *(const int2*)(ar + 16);
            int2 d2 = *(const int2*)(ar + 32);
            int2 d3 = *(const int2*)(ar + 48);
            v8i a;
            a[0] = d0.x; a[1] = d0.y; a[2] = d1.x; a[3] = d1.y;
            a[4] = d2.x; a[5] = d2.y; a[6] = d3.x; a[7] = d3.y;
            afrag[i] = a;
        }
#pragma unroll
        for (int j = 0; j < 4; ++j) {
            const int8_t* br = &ldsB[p][(waveO + j * 16 + ln) * LDSS + h * 8];
            int2 q0 = *(const int2*)(br);
            int2 q1 = *(const int2*)(br + 16);
            int2 q2 = *(const int2*)(br + 32);
            int2 q3 = *(const int2*)(br + 48);
            v8i b;
            b[0] = q0.x; b[1] = q0.y; b[2] = q1.x; b[3] = q1.y;
            b[4] = q2.x; b[5] = q2.y; b[6] = q3.x; b[7] = q3.y;
            bfrag[j] = b;
        }
#pragma unroll
        for (int i = 0; i < 2; ++i)
#pragma unroll
            for (int j = 0; j < 4; ++j)
                acc[i][j] = __builtin_amdgcn_wmma_i32_16x16x64_iu8(
                     true, afrag[i],  true, bfrag[j],
                    acc[i][j],  false,  false);

        __syncthreads();
    }

#pragma unroll
    for (int j = 0; j < 4; ++j) {
        int out = tileO + waveO + j * 16 + ln;
        float rs  = row_scales[out];
        float bo  = bias[out];
        int   wrs = w_rowsum[out];
#pragma unroll
        for (int i = 0; i < 2; ++i) {
#pragma unroll
            for (int r = 0; r < 8; ++r) {
                int token = tileT + waveT + i * 16 + r + h * 8;
                float cs = col_scales[token];
                int   zp = zps[token];
                float y = rs * cs * (float)(acc[i][j][r] - zp * wrs) + bo;
                Y[(size_t)token * DOUT + out] = y;
            }
        }
    }
}

extern "C" void kernel_launch(void* const* d_in, const int* in_sizes, int n_in,
                              void* d_out, int out_size, void* d_ws, size_t ws_size,
                              hipStream_t stream) {
    const float* x    = (const float*)d_in[0];
    const float* W    = (const float*)d_in[1];
    const float* bias = (const float*)d_in[2];
    const float* cmin = (const float*)d_in[3];
    const float* cmax = (const float*)d_in[4];
    float* out = (float*)d_out;

    char* ws = (char*)d_ws;
    int8_t* Wq         = (int8_t*)ws;
    int8_t* Xq         = (int8_t*)(ws + (size_t)16 * 1024 * 1024);
    char*   meta       = ws + (size_t)48 * 1024 * 1024;
    float*  row_scales = (float*)(meta);
    int*    w_rowsum   = (int*)(meta + 16 * 1024);
    float*  col_scales = (float*)(meta + 32 * 1024);
    int*    zps        = (int*)(meta + 64 * 1024);

    wquant_kernel<<<DOUT, 256, 0, stream>>>(W, Wq, row_scales, w_rowsum);
    xquant_kernel<<<NTOK, 256, 0, stream>>>(x, cmin, cmax, Xq, col_scales, zps);
    gemm_kernel<<<dim3(NTOK / 128, DOUT / 128), 256, 0, stream>>>(
        Xq, Wq, row_scales, w_rowsum, col_scales, zps, bias, out);
}
